// MultiHeadAttention_10118942950053
// MI455X (gfx1250) — matrix operand order repaired and checked by reading, not compiled, not run on hardware
//
#include <hip/hip_runtime.h>
#include <hip/hip_bf16.h>
#include <math.h>

// ---------------------------------------------------------------------------
// Memory-augmented MHA + residual LayerNorm for gfx1250 (MI455X), wave32 WMMA.
// D=1024, H=16, DK=64, B=4, N=1024, M=64 memory tokens, N+M=1088.
// All GEMMs / attention einsums via v_wmma_f32_16x16x32_f16.
//  - Inputs and weights are pre-converted to f16 once; GEMM staging is pure
//    b128 copies (no conversion in the hot loop), BK=64 -> 8 WMMAs/barrier.
//  - Attention is computed fully transposed (S^T = K*Q^T, O^T = V^T*P^T) so
//    softmax is per-lane (one shfl_xor(16) per statistic) and all LDS/global
//    traffic is packed 128-bit.
// ---------------------------------------------------------------------------

typedef __attribute__((ext_vector_type(16))) _Float16 v16h;
typedef __attribute__((ext_vector_type(8)))  _Float16 v8h;
typedef __attribute__((ext_vector_type(4)))  _Float16 v4h;
typedef __attribute__((ext_vector_type(8)))  float    v8f;
typedef __attribute__((ext_vector_type(4)))  float    v4f;

union HFrag { v16h v; v8h h[2]; };

#define WMMA16(a, b, c) \
  __builtin_amdgcn_wmma_f32_16x16x32_f16(false, (a), false, (b), (short)0, (c), false, false)

#define D_MODEL 1024
#define N_SEQ   1024
#define N_BATCH 4
#define N_HEADS 16
#define DK_HEAD 64
#define N_MEM   64
#define NKV     1088   // N_SEQ + N_MEM
#define NEGINF  (-__builtin_inff())

// ---------------------------------------------------------------------------
// Kernel 0: f32 -> f16 elementwise conversion (vectorized 4-wide)
// ---------------------------------------------------------------------------
__global__ __launch_bounds__(256) void cvt_f16(const float* __restrict__ src,
                                               _Float16* __restrict__ dst,
                                               int n4) {
  int i = blockIdx.x * 256 + threadIdx.x;
  if (i >= n4) return;
  v4f x = *(const v4f*)(src + (size_t)i * 4);
  v4h y;
  y[0] = (_Float16)x[0]; y[1] = (_Float16)x[1];
  y[2] = (_Float16)x[2]; y[3] = (_Float16)x[3];
  *(v4h*)(dst + (size_t)i * 4) = y;
}

// ---------------------------------------------------------------------------
// Kernel 1: fill memory rows of Kfull (f16) and memory columns of V-transposed
//           m_k * sqrt(DK)=8 ; m_v * sqrt(M)=8
// ---------------------------------------------------------------------------
__global__ __launch_bounds__(256) void fill_mem(const float* __restrict__ mk,
                                                const float* __restrict__ mv,
                                                _Float16* __restrict__ Kfull,
                                                _Float16* __restrict__ VT) {
  int idx = blockIdx.x * 256 + threadIdx.x;       // B*M*D = 262144
  if (idx >= N_BATCH * N_MEM * D_MODEL) return;
  int d = idx & (D_MODEL - 1);
  int m = (idx >> 10) & (N_MEM - 1);
  int b = idx >> 16;
  float kv = mk[m * D_MODEL + d] * 8.0f;
  float vv = mv[m * D_MODEL + d] * 8.0f;
  Kfull[(size_t)b * NKV * D_MODEL + (size_t)(N_SEQ + m) * D_MODEL + d] = (_Float16)kv;
  VT[(size_t)b * D_MODEL * NKV + (size_t)d * NKV + (N_SEQ + m)] = (_Float16)vv;
}

// ---------------------------------------------------------------------------
// Kernel 2: GEMM  C[m,n] = sum_k A[m,k] * W[n,k] + bias[n]   (A, W in f16)
//   M = 4096, N = 1024, K = 1024.  Block: 256 threads (8 waves),
//   block tile 128x64, BK = 64.  Each wave computes 32x32 (4 accumulators,
//   8 WMMAs per staged slab).
//   MODE 0: store f16 row-major (Q buffer, stride 1024)
//   MODE 1: store f16 into Kfull [B,1088,1024] (rows 0..1023 of each batch)
//   MODE 2: store f16 transposed into VT [B,1024,1088]
//   MODE 3: store f32 row-major (output projection)
// ---------------------------------------------------------------------------
template <int MODE>
__global__ __launch_bounds__(256) void gemm_wmma(const _Float16* __restrict__ A,
                                                 const _Float16* __restrict__ W,
                                                 const float* __restrict__ bias,
                                                 void* __restrict__ Cptr) {
  __shared__ _Float16 Alds[128][72];  // 72-half row stride: 144B, 16B aligned
  __shared__ _Float16 Blds[64][72];

  const int t    = threadIdx.x;
  const int lane = t & 31;
  const int w    = t >> 5;            // wave 0..7
  const int lm   = lane & 15;
  const int hi   = lane >> 4;
  const int mrow = (w >> 1) * 32;     // 0,32,64,96
  const int ncol = (w & 1) * 32;      // 0,32
  const int m0   = blockIdx.x * 128;
  const int n0   = blockIdx.y * 64;

  v8f acc00 = {}, acc01 = {}, acc10 = {}, acc11 = {};

  for (int k0 = 0; k0 < D_MODEL; k0 += 64) {
    // ---- stage A slab (128x64 halves) : 4 b128 copies per thread ----
    #pragma unroll
    for (int rep = 0; rep < 4; ++rep) {
      const int chunk = t + rep * 256;          // 1024 chunks of 8 halves
      const int i = chunk >> 3, c = (chunk & 7) * 8;
      v8h x = *(const v8h*)(A + (size_t)(m0 + i) * D_MODEL + k0 + c);
      *(v8h*)&Alds[i][c] = x;
    }
    // ---- stage B slab: W rows n0..n0+63, cols k0..k0+63 ----
    #pragma unroll
    for (int rep = 0; rep < 2; ++rep) {
      const int chunk = t + rep * 256;          // 512 chunks of 8 halves
      const int i = chunk >> 3, c = (chunk & 7) * 8;
      v8h x = *(const v8h*)(W + (size_t)(n0 + i) * D_MODEL + k0 + c);
      *(v8h*)&Blds[i][c] = x;
    }
    __syncthreads();

    // ---- two K=32 sub-steps, 4 WMMAs each ----
    #pragma unroll
    for (int kk = 0; kk < 64; kk += 32) {
      HFrag af0, af1, bf0, bf1;
      af0.h[0] = *(const v8h*)&Alds[mrow + lm][kk + hi * 8];
      af0.h[1] = *(const v8h*)&Alds[mrow + lm][kk + 16 + hi * 8];
      af1.h[0] = *(const v8h*)&Alds[mrow + 16 + lm][kk + hi * 8];
      af1.h[1] = *(const v8h*)&Alds[mrow + 16 + lm][kk + 16 + hi * 8];
      bf0.h[0] = *(const v8h*)&Blds[ncol + lm][kk + hi * 8];
      bf0.h[1] = *(const v8h*)&Blds[ncol + lm][kk + 16 + hi * 8];
      bf1.h[0] = *(const v8h*)&Blds[ncol + 16 + lm][kk + hi * 8];
      bf1.h[1] = *(const v8h*)&Blds[ncol + 16 + lm][kk + 16 + hi * 8];
      acc00 = WMMA16(af0.v, bf0.v, acc00);
      acc01 = WMMA16(af0.v, bf1.v, acc01);
      acc10 = WMMA16(af1.v, bf0.v, acc10);
      acc11 = WMMA16(af1.v, bf1.v, acc11);
    }
    __syncthreads();
  }

  // ---- epilogue ----
  #pragma unroll
  for (int s = 0; s < 2; ++s) {
    #pragma unroll
    for (int a = 0; a < 2; ++a) {
      const int col = n0 + ncol + a * 16 + lm;
      const float bv = bias[col];
      v8f acc = s ? (a ? acc11 : acc10) : (a ? acc01 : acc00);
      #pragma unroll
      for (int r = 0; r < 8; ++r) {
        const int grow = m0 + mrow + s * 16 + r + hi * 8;   // global M row
        const float val = acc[r] + bv;
        if (MODE == 0) {
          ((_Float16*)Cptr)[(size_t)grow * D_MODEL + col] = (_Float16)val;
        } else if (MODE == 1) {
          const int b = grow >> 10, i = grow & (N_SEQ - 1);
          ((_Float16*)Cptr)[(size_t)b * NKV * D_MODEL + (size_t)i * D_MODEL + col] =
              (_Float16)val;
        } else if (MODE == 2) {
          const int b = grow >> 10, i = grow & (N_SEQ - 1);
          ((_Float16*)Cptr)[(size_t)b * D_MODEL * NKV + (size_t)col * NKV + i] =
              (_Float16)val;
        } else {
          ((float*)Cptr)[(size_t)grow * D_MODEL + col] = val;
        }
      }
    }
  }
}

// ---------------------------------------------------------------------------
// Kernel 3: transposed flash attention over 1088 keys (last 64 = memory,
// never masked).  Grid: B*H*(N/64) = 1024 blocks, 128 threads (4 waves);
// each wave owns a 16-row q tile.  Per 32-key step:
//   S^T = K_tile * Q^T      (4 WMMAs; C-frag: lane <-> q row, regs <-> keys)
//   per-lane softmax (one shfl_xor(16) for max, one for sum; mask via ballot)
//   P^T -> LDS as two packed b128 stores
//   O^T += V^T_tile * P^T   (4 WMMAs; accumulator rescale is per-lane)
// Mask semantics match the reference: mask != 0 -> -inf.
// ---------------------------------------------------------------------------
__global__ __launch_bounds__(128) void attn_kernel(const _Float16* __restrict__ Q,
                                                   const _Float16* __restrict__ Kfull,
                                                   const _Float16* __restrict__ VT,
                                                   const int* __restrict__ amask,
                                                   _Float16* __restrict__ AO) {
  __shared__ _Float16 plds[4][16][40];   // per-wave P^T tile, 80B rows

  const int t    = threadIdx.x;
  const int lane = t & 31;
  const int w    = t >> 5;
  const int lm   = lane & 15;
  const int hi   = lane >> 4;

  const int blk = blockIdx.x;
  const int b   = blk >> 8;
  const int h   = (blk >> 4) & 15;
  const int qt  = blk & 15;
  const int qrow = qt * 64 + w * 16;     // q row base for this wave

  // Q^T B-fragments (lane n = q column), fixed for the whole kernel.
  const _Float16* qp =
      Q + (size_t)(b * N_SEQ + qrow + lm) * D_MODEL + h * DK_HEAD;
  HFrag qb0, qb1;
  qb0.h[0] = *(const v8h*)(qp + hi * 8);
  qb0.h[1] = *(const v8h*)(qp + 16 + hi * 8);
  qb1.h[0] = *(const v8h*)(qp + 32 + hi * 8);
  qb1.h[1] = *(const v8h*)(qp + 48 + hi * 8);

  float m_run = NEGINF, l_run = 0.0f;
  v8f ot0 = {}, ot1 = {}, ot2 = {}, ot3 = {};   // O^T: regs <-> d, lane <-> q

  const _Float16* kbase = Kfull + (size_t)b * NKV * D_MODEL + h * DK_HEAD;
  const _Float16* vbase =
      VT + (size_t)b * D_MODEL * NKV + (size_t)(h * DK_HEAD) * NKV;
  const int* mp = amask + b * N_SEQ;

  for (int j = 0; j < NKV; j += 32) {
    if (j + 32 < NKV)
      __builtin_prefetch(kbase + (size_t)(j + 32 + lm) * D_MODEL, 0, 0);

    // key-mask bitmap for these 32 keys (key j+lane <-> lane)
    const int kidx = j + lane;
    const int mval = (kidx < N_SEQ) ? mp[kidx] : 0;
    const unsigned int mbits = (unsigned int)__ballot(mval != 0);

    // ---- S^T: two 16-key subtiles (A-frag = K rows, B-frag = Q^T) ----
    v8f s0 = {}, s1 = {};
    #pragma unroll
    for (int st = 0; st < 2; ++st) {
      const _Float16* kp = kbase + (size_t)(j + st * 16 + lm) * D_MODEL;
      HFrag ka0, ka1;
      ka0.h[0] = *(const v8h*)(kp + hi * 8);
      ka0.h[1] = *(const v8h*)(kp + 16 + hi * 8);
      ka1.h[0] = *(const v8h*)(kp + 32 + hi * 8);
      ka1.h[1] = *(const v8h*)(kp + 48 + hi * 8);
      v8f s = {};
      s = WMMA16(ka0.v, qb0.v, s);
      s = WMMA16(ka1.v, qb1.v, s);
      if (st) s1 = s; else s0 = s;
    }

    // ---- scale + mask: reg r of tile st holds key st*16 + r + 8*hi ----
    #pragma unroll
    for (int r = 0; r < 8; ++r) {
      s0[r] = ((mbits >> (r + 8 * hi)) & 1u)      ? NEGINF : s0[r] * 0.125f;
      s1[r] = ((mbits >> (16 + r + 8 * hi)) & 1u) ? NEGINF : s1[r] * 0.125f;
    }

    // ---- per-lane online softmax over keys ----
    float tm = s0[0];
    #pragma unroll
    for (int r = 1; r < 8; ++r) tm = fmaxf(tm, s0[r]);
    #pragma unroll
    for (int r = 0; r < 8; ++r) tm = fmaxf(tm, s1[r]);
    tm = fmaxf(tm, __shfl_xor(tm, 16));
    const float mn = fmaxf(m_run, tm);
    const float c = (m_run == mn) ? 1.0f : __expf(m_run - mn);

    float rs = 0.0f;
    v8h ph0 = {}, ph1 = {};
    #pragma unroll
    for (int r = 0; r < 8; ++r) {
      float p0 = (s0[r] < -1e37f) ? 0.0f : __expf(s0[r] - mn);
      float p1 = (s1[r] < -1e37f) ? 0.0f : __expf(s1[r] - mn);
      rs += p0 + p1;
      ph0[r] = (_Float16)p0;
      ph1[r] = (_Float16)p1;
    }
    rs += __shfl_xor(rs, 16);
    l_run = l_run * c + rs;
    m_run = mn;

    // per-lane accumulator rescale (c is per q-row == per lane)
    #pragma unroll
    for (int r = 0; r < 8; ++r) {
      ot0[r] *= c; ot1[r] *= c; ot2[r] *= c; ot3[r] *= c;
    }

    // ---- P^T -> LDS: row q = lm, keys st*16 + 8*hi + 0..7 (packed) ----
    *(v8h*)&plds[w][lm][8 * hi]      = ph0;
    *(v8h*)&plds[w][lm][16 + 8 * hi] = ph1;

    // P^T B-fragment (lane n = q column, K = keys)
    HFrag pb;
    pb.h[0] = *(const v8h*)&plds[w][lm][hi * 8];
    pb.h[1] = *(const v8h*)&plds[w][lm][16 + hi * 8];

    // ---- O^T += V^T * P^T (A-frag = V^T rows: lane m = d) ----
    #pragma unroll
    for (int dt = 0; dt < 4; ++dt) {
      const _Float16* vp = vbase + (size_t)(dt * 16 + lm) * NKV + j;
      HFrag va;
      va.h[0] = *(const v8h*)(vp + hi * 8);
      va.h[1] = *(const v8h*)(vp + 16 + hi * 8);
      if (dt == 0)      ot0 = WMMA16(va.v, pb.v, ot0);
      else if (dt == 1) ot1 = WMMA16(va.v, pb.v, ot1);
      else if (dt == 2) ot2 = WMMA16(va.v, pb.v, ot2);
      else              ot3 = WMMA16(va.v, pb.v, ot3);
    }
  }

  // ---- normalize (per-lane) and store packed b128 per d-subtile ----
  const float inv = 1.0f / l_run;
  v8h r0 = {}, r1 = {}, r2 = {}, r3 = {};
  #pragma unroll
  for (int r = 0; r < 8; ++r) {
    r0[r] = (_Float16)(ot0[r] * inv);
    r1[r] = (_Float16)(ot1[r] * inv);
    r2[r] = (_Float16)(ot2[r] * inv);
    r3[r] = (_Float16)(ot3[r] * inv);
  }
  _Float16* op = AO + (size_t)(b * N_SEQ + qrow + lm) * D_MODEL +
                 h * DK_HEAD + 8 * hi;
  *(v8h*)(op + 0)  = r0;   // d = 0*16 + 8*hi + r
  *(v8h*)(op + 16) = r1;
  *(v8h*)(op + 32) = r2;
  *(v8h*)(op + 48) = r3;
}

// ---------------------------------------------------------------------------
// Kernel 4: residual + LayerNorm.  out = LN(queries + proj) * gamma + beta
// One 256-thread block per row (4096 rows, 1024 cols).
// ---------------------------------------------------------------------------
__global__ __launch_bounds__(256) void ln_kernel(const float* __restrict__ q,
                                                 const float* __restrict__ proj,
                                                 const float* __restrict__ gamma,
                                                 const float* __restrict__ beta,
                                                 float* __restrict__ out) {
  __shared__ float rs[8], rs2[8];
  const int row = blockIdx.x;
  const int t = threadIdx.x;
  const float* qa = q + (size_t)row * D_MODEL;
  const float* ta = proj + (size_t)row * D_MODEL;

  float x[4], s = 0.0f, s2 = 0.0f;
  #pragma unroll
  for (int i = 0; i < 4; ++i) {
    const int c = t + i * 256;
    const float v = qa[c] + ta[c];
    x[i] = v; s += v; s2 += v * v;
  }
  #pragma unroll
  for (int d = 1; d < 32; d <<= 1) {
    s += __shfl_xor(s, d);
    s2 += __shfl_xor(s2, d);
  }
  if ((t & 31) == 0) { rs[t >> 5] = s; rs2[t >> 5] = s2; }
  __syncthreads();
  s = 0.0f; s2 = 0.0f;
  #pragma unroll
  for (int i = 0; i < 8; ++i) { s += rs[i]; s2 += rs2[i]; }

  const float mu = s * (1.0f / D_MODEL);
  const float var = s2 * (1.0f / D_MODEL) - mu * mu;
  const float rinv = rsqrtf(var + 1e-5f);
  #pragma unroll
  for (int i = 0; i < 4; ++i) {
    const int c = t + i * 256;
    out[(size_t)row * D_MODEL + c] = (x[i] - mu) * rinv * gamma[c] + beta[c];
  }
}

// ---------------------------------------------------------------------------
// Launch
// ---------------------------------------------------------------------------
extern "C" void kernel_launch(void* const* d_in, const int* in_sizes, int n_in,
                              void* d_out, int out_size, void* d_ws, size_t ws_size,
                              hipStream_t stream) {
  const float* queries = (const float*)d_in[0];
  const float* keys    = (const float*)d_in[1];
  const float* values  = (const float*)d_in[2];
  const int*   amask   = (const int*)d_in[3];
  const float* Wq = (const float*)d_in[4];
  const float* bq = (const float*)d_in[5];
  const float* Wk = (const float*)d_in[6];
  const float* bk = (const float*)d_in[7];
  const float* Wv = (const float*)d_in[8];
  const float* bv = (const float*)d_in[9];
  const float* Wo = (const float*)d_in[10];
  const float* bo = (const float*)d_in[11];
  const float* mk = (const float*)d_in[12];
  const float* mv = (const float*)d_in[13];
  const float* gamma = (const float*)d_in[14];
  const float* beta  = (const float*)d_in[15];
  float* out = (float*)d_out;

  // workspace layout
  char* ws = (char*)d_ws;
  const size_t bX  = (size_t)N_BATCH * N_SEQ * D_MODEL * 2;   // 8 MB per tensor
  const size_t bW  = (size_t)D_MODEL * D_MODEL * 2;           // 2 MB per weight
  const size_t bKF = (size_t)N_BATCH * NKV * D_MODEL * 2;     // 8.9 MB
  const size_t bVT = (size_t)N_BATCH * D_MODEL * NKV * 2;     // 8.9 MB

  size_t off = 0;
  _Float16* Xq16 = (_Float16*)(ws + off); off += bX;
  _Float16* Xk16 = (_Float16*)(ws + off); off += bX;
  _Float16* Xv16 = (_Float16*)(ws + off); off += bX;
  _Float16* Wq16 = (_Float16*)(ws + off); off += bW;
  _Float16* Wk16 = (_Float16*)(ws + off); off += bW;
  _Float16* Wv16 = (_Float16*)(ws + off); off += bW;
  _Float16* Wo16 = (_Float16*)(ws + off); off += bW;
  _Float16* Qbf   = (_Float16*)(ws + off); off += bX;
  _Float16* Kfull = (_Float16*)(ws + off); off += bKF;
  _Float16* VT    = (_Float16*)(ws + off); off += bVT;
  _Float16* AO    = (_Float16*)(ws + off); off += bX;
  float*    TMP   = (float*)(ws + off);

  // 0) pre-convert activations + weights to f16 (once)
  const int nX4 = N_BATCH * N_SEQ * D_MODEL / 4;   // 1,048,576
  const int nW4 = D_MODEL * D_MODEL / 4;           // 262,144
  cvt_f16<<<nX4 / 256, 256, 0, stream>>>(queries, Xq16, nX4);
  cvt_f16<<<nX4 / 256, 256, 0, stream>>>(keys,    Xk16, nX4);
  cvt_f16<<<nX4 / 256, 256, 0, stream>>>(values,  Xv16, nX4);
  cvt_f16<<<nW4 / 256, 256, 0, stream>>>(Wq, Wq16, nW4);
  cvt_f16<<<nW4 / 256, 256, 0, stream>>>(Wk, Wk16, nW4);
  cvt_f16<<<nW4 / 256, 256, 0, stream>>>(Wv, Wv16, nW4);
  cvt_f16<<<nW4 / 256, 256, 0, stream>>>(Wo, Wo16, nW4);

  // 1) memory tokens into Kfull / VT
  fill_mem<<<1024, 256, 0, stream>>>(mk, mv, Kfull, VT);

  // 2) Q/K/V projections (f16 in, f16 out; V stored transposed)
  dim3 ggrid(32, 16), gblk(256);
  gemm_wmma<0><<<ggrid, gblk, 0, stream>>>(Xq16, Wq16, bq, (void*)Qbf);
  gemm_wmma<1><<<ggrid, gblk, 0, stream>>>(Xk16, Wk16, bk, (void*)Kfull);
  gemm_wmma<2><<<ggrid, gblk, 0, stream>>>(Xv16, Wv16, bv, (void*)VT);

  // 3) attention (transposed flash)
  attn_kernel<<<1024, 128, 0, stream>>>(Qbf, Kfull, VT, amask, AO);

  // 4) output projection (f16 in, f32 out)
  gemm_wmma<3><<<ggrid, gblk, 0, stream>>>(AO, Wo16, bo, (void*)TMP);

  // 5) residual + LayerNorm
  ln_kernel<<<4096, 256, 0, stream>>>(queries, TMP, gamma, beta, out);
}
